// Encoder_6201932776092
// MI455X (gfx1250) — matrix operand order repaired and checked by reading, not compiled, not run on hardware
//
#include <hip/hip_runtime.h>
#include <hip/hip_bf16.h>

// ---------------------------------------------------------------------------
// Encoder: embed -> 3x(conv1d K=5 + BN + ReLU) -> biLSTM(256) -> [B,T,512] f32
// All GEMM-shaped math via v_wmma_f32_16x16x32_f16 (wave32, CDNA5).
// v4: fix async-to-LDS builtin arg types: signature is
//     (int4 addrspace(1)*, int4 addrspace(3)*, imm int offset, imm int cpol)
//     (revealed by round-3 diagnostic). Rest unchanged from v3.
// ---------------------------------------------------------------------------

typedef __attribute__((ext_vector_type(16))) _Float16 v16h;
typedef __attribute__((ext_vector_type(8)))  _Float16 v8h;
typedef __attribute__((ext_vector_type(8)))  float    v8f;

#define EMB    512
#define FILT   512
#define KW     5
#define UNITS  256
#define BATCH  64
#define TLEN   512
#define ROWS   (BATCH * TLEN)     // 32768 (b*T + t) rows
#define BN_EPS 1e-3f

#if defined(__AMDGCN__) && __has_builtin(__builtin_amdgcn_global_load_async_to_lds_b128)
#define HAVE_ASYNC_LDS 1
typedef int v4i_b128 __attribute__((vector_size(16)));
typedef __attribute__((address_space(1))) v4i_b128* as1_b128;
typedef __attribute__((address_space(3))) v4i_b128* as3_b128;
#else
#define HAVE_ASYNC_LDS 0
#endif

// ---------------------------------------------------------------------------
// WMMA helper
// ---------------------------------------------------------------------------
__device__ __forceinline__ v8f wmma16x16x32(v16h a, v16h b, v8f c) {
  return __builtin_amdgcn_wmma_f32_16x16x32_f16(
      /*neg_a=*/false, a, /*neg_b=*/false, b,
      /*c_mod=*/(short)0, c, /*reuse_a=*/false, /*reuse_b=*/false);
}

// A-fragment (16x32 f16, MxK). ISA layout: lane L -> row m=L%16, half=L/16.
// elements e=0..7  hold K = kbase + 8*half + e          (VGPR0..3)
// elements e=8..15 hold K = kbase + 16 + 8*half + (e-8) (VGPR4..7)
// => two contiguous 16-byte loads per lane.
__device__ __forceinline__ v16h load_a_frag(const _Float16* row, int kbase, int half) {
  v8h lo = *(const v8h*)(row + kbase + half * 8);
  v8h hi = *(const v8h*)(row + kbase + 16 + half * 8);
  v16h a;
#pragma unroll
  for (int i = 0; i < 8; ++i) { a[i] = lo[i]; a[i + 8] = hi[i]; }
  return a;
}

// ---------------------------------------------------------------------------
// Weight packing: B-fragment layout (32x16 f16, KxN).
// lane L -> column n = nt*16 + L%16, half = L/16.
// element e holds K = kc*32 + 8*half + (e&7) + 16*(e>>3) (matches A-fragment layout).
// Packed as [kc][nt][lane][16 halves] contiguous -> one 32B load per fragment.
// ---------------------------------------------------------------------------
__global__ void pack_b_kernel(const float* __restrict__ src, _Float16* __restrict__ dst,
                              int kchunks, int ntiles, int N) {
  long long total = (long long)kchunks * ntiles * 32 * 16;
  long long i = (long long)blockIdx.x * blockDim.x + threadIdx.x;
  if (i >= total) return;
  int e = (int)(i & 15);
  long long r = i >> 4;
  int lane = (int)(r & 31); r >>= 5;
  int nt = (int)(r % ntiles);
  int kc = (int)(r / ntiles);
  int half = lane >> 4;
  int n = nt * 16 + (lane & 15);
  int k = kc * 32 + 8 * half + (e & 7) + 16 * (e >> 3);
  dst[i] = (_Float16)src[(long long)k * N + n];
}

// ---------------------------------------------------------------------------
// Embedding gather -> f16 activations, [ROWS, EMB], row = b*T + t
// ---------------------------------------------------------------------------
__global__ void embed_kernel(const int* __restrict__ x, const float* __restrict__ emb,
                             _Float16* __restrict__ X0) {
  int i = blockIdx.x * blockDim.x + threadIdx.x;   // over ROWS * (EMB/8)
  int row = i >> 6;        // EMB/8 = 64 vec8 per row
  int g = (i & 63) * 8;
  int tok = x[row];
  const float* src = emb + (long long)tok * EMB + g;
  v8h o;
#pragma unroll
  for (int j = 0; j < 8; ++j) o[j] = (_Float16)src[j];
  *(v8h*)(X0 + (long long)row * EMB + g) = o;
}

// ---------------------------------------------------------------------------
// Fold BN + conv bias into per-filter alpha/beta:
//   y = relu( conv * alpha + beta ),  alpha = g*rsqrt(v+eps),
//   beta = b_bn - mean*alpha + bias*alpha
// ---------------------------------------------------------------------------
__global__ void bnfold_kernel(const float* __restrict__ conv_b, const float* __restrict__ gamma,
                              const float* __restrict__ beta, const float* __restrict__ mean,
                              const float* __restrict__ var, float* __restrict__ alpha,
                              float* __restrict__ betap) {
  int i = blockIdx.x * blockDim.x + threadIdx.x;
  if (i >= 3 * FILT) return;
  float s = gamma[i] * rsqrtf(var[i] + BN_EPS);
  alpha[i] = s;
  betap[i] = beta[i] - mean[i] * s + conv_b[i] * s;
}

// ---------------------------------------------------------------------------
// Conv layer as implicit GEMM: M=ROWS, N=FILT, K=KW*EMB=2560.
// Block (256 thr, 8 waves) computes 32 rows x 256 filters:
//   - stages 36 activation rows (halo of K=5) in LDS once (bank-staggered,
//     async-to-LDS when available)
//   - wave = (mi in 0..1, ni in 0..3): 16 rows x 4 n-tiles, 4 accumulators,
//     each A fragment (from LDS) reused for 4 WMMAs.
// grid = (ROWS/32, FILT/256) = (1024, 2)
// ---------------------------------------------------------------------------
#define CONV_ROWS   36
#define CONV_STRIDE 520   // halves; 1040B row stride: 16B aligned, 4-bank stagger

__global__ void __launch_bounds__(256) conv_kernel(
    const _Float16* __restrict__ X, const _Float16* __restrict__ Wp,
    const float* __restrict__ alpha, const float* __restrict__ betap,
    _Float16* __restrict__ Y) {
  __shared__ __align__(16) _Float16 atile[CONV_ROWS * CONV_STRIDE];  // ~36.6 KB
  const int NT = FILT / 16;        // 32 global n-tiles
  const int KC = KW * EMB / 32;    // 80 k-chunks
  const int mb = blockIdx.x;       // 32 consecutive (b*T+t) rows
  const int nb = blockIdx.y;       // 16 n-tiles per block
  const int tid = threadIdx.x;
  const int g0 = mb * 32;
  const int b = g0 >> 9, t0 = g0 & 511;   // T=512 divides 32-row groups

  // cooperative stage of rows tsrc = t0-2 .. t0+33 (zero pad outside [0,T))
#pragma unroll
  for (int k = 0; k < 9; ++k) {
    int idx = tid + k * 256;       // 36 rows * 64 chunks of 8 halves = 2304
    int rrow = idx >> 6;
    int c8 = (idx & 63) * 8;
    int ts = t0 - 2 + rrow;
    _Float16* ldst = atile + rrow * CONV_STRIDE + c8;
    if (ts >= 0 && ts < TLEN) {
      const _Float16* gsrc = X + ((long long)(b << 9) + ts) * EMB + c8;
#if HAVE_ASYNC_LDS
      __builtin_amdgcn_global_load_async_to_lds_b128(
          (as1_b128)(void*)const_cast<_Float16*>(gsrc),
          (as3_b128)(void*)ldst, 0, 0);
#else
      *(v8h*)ldst = *(const v8h*)gsrc;
#endif
    } else {
      *(v8h*)ldst = (v8h){};
    }
  }
#if HAVE_ASYNC_LDS
#if __has_builtin(__builtin_amdgcn_s_wait_asynccnt)
  __builtin_amdgcn_s_wait_asynccnt(0);
#else
  asm volatile("s_wait_asynccnt 0x0" ::: "memory");
#endif
#endif
  __syncthreads();

  const int wave = tid >> 5, lane = tid & 31, half = lane >> 4, l15 = lane & 15;
  const int mi = wave & 1;         // m-tile within block
  const int ni = wave >> 1;        // n-group (4 tiles)
  const int ntb = nb * 16 + ni * 4;
  const v16h* wfr = (const v16h*)Wp;

  v8f acc0 = {}, acc1 = {}, acc2 = {}, acc3 = {};
  for (int kc = 0; kc < KC; ++kc) {
    int tap = kc >> 4;                       // EMB/32 = 16 chunks per tap
    int c0  = (kc & 15) * 32;
    const _Float16* arow = atile + (mi * 16 + l15 + tap) * CONV_STRIDE;
    v16h a = load_a_frag(arow, c0, half);
    const v16h* bp = wfr + ((long long)kc * NT + ntb) * 32 + lane;
    // group the 8 b128 weight loads, then 4 back-to-back WMMAs (shared A)
    v16h b0 = bp[0];
    v16h b1 = bp[32];
    v16h b2 = bp[64];
    v16h b3 = bp[96];
    __builtin_prefetch((const void*)(bp + (long long)NT * 32), 0, 3);
    acc0 = wmma16x16x32(a, b0, acc0);
    acc1 = wmma16x16x32(a, b1, acc1);
    acc2 = wmma16x16x32(a, b2, acc2);
    acc3 = wmma16x16x32(a, b3, acc3);
  }

  v8f accs[4] = {acc0, acc1, acc2, acc3};
#pragma unroll
  for (int j = 0; j < 4; ++j) {
    int n = (ntb + j) * 16 + l15;
    float al = alpha[n], be = betap[n];
#pragma unroll
    for (int r = 0; r < 8; ++r) {
      int m = g0 + mi * 16 + r + 8 * half;   // C/D layout: lanes16-31 -> M+8
      float y = accs[j][r] * al + be;
      y = y > 0.f ? y : 0.f;
      Y[(long long)m * FILT + n] = (_Float16)y;
    }
  }
}

// ---------------------------------------------------------------------------
// LSTM input projection: Z = X @ Wk + b;  M=ROWS, N=4U=1024, K=512. f16 out.
// Wave computes 16 rows x 4 n-tiles (A fragment reused 4x).
// ---------------------------------------------------------------------------
__global__ void __launch_bounds__(256) proj_kernel(
    const _Float16* __restrict__ X, const _Float16* __restrict__ Wp,
    const float* __restrict__ bias, _Float16* __restrict__ Z) {
  const int NT = 64;                // 1024/16 n-tiles
  const int KC = EMB / 32;          // 16
  int wave = blockIdx.x * (blockDim.x >> 5) + (threadIdx.x >> 5);
  int lane = threadIdx.x & 31, half = lane >> 4, l15 = lane & 15;
  int ng = wave & 15;               // n-group of 4 tiles
  int mt = wave >> 4;               // 0..2047
  int grow = mt * 16 + l15;
  const v16h* wfr = (const v16h*)Wp;
  v8f acc0 = {}, acc1 = {}, acc2 = {}, acc3 = {};
  for (int kc = 0; kc < KC; ++kc) {
    v16h a = load_a_frag(X + (long long)grow * EMB, kc * 32, half);
    const v16h* bp = wfr + ((long long)kc * NT + ng * 4) * 32 + lane;
    v16h b0 = bp[0];
    v16h b1 = bp[32];
    v16h b2 = bp[64];
    v16h b3 = bp[96];
    __builtin_prefetch((const void*)(bp + (long long)NT * 32), 0, 3);
    acc0 = wmma16x16x32(a, b0, acc0);
    acc1 = wmma16x16x32(a, b1, acc1);
    acc2 = wmma16x16x32(a, b2, acc2);
    acc3 = wmma16x16x32(a, b3, acc3);
  }
  v8f accs[4] = {acc0, acc1, acc2, acc3};
#pragma unroll
  for (int j = 0; j < 4; ++j) {
    int n = (ng * 4 + j) * 16 + l15;
    float bz = bias[n];
#pragma unroll
    for (int r = 0; r < 8; ++r) {
      int m = mt * 16 + r + 8 * half;
      Z[(long long)m * (4 * UNITS) + n] = (_Float16)(accs[j][r] + bz);
    }
  }
}

// ---------------------------------------------------------------------------
// Bidirectional LSTM recurrence. blockIdx.x = direction (0 fwd, 1 bwd).
// 1024 threads = 32 waves; wave -> (batch tile mt in 0..3, u-tiles ut0, ut0+1).
// Each wave computes all 4 gates for its cells => in-register gate fusion.
// h kept f16 in LDS (A fragments); c in registers; 2 barriers per step.
// ---------------------------------------------------------------------------
__global__ void __launch_bounds__(1024, 1) lstm_kernel(
    const _Float16* __restrict__ Zf, const _Float16* __restrict__ Zb,
    const _Float16* __restrict__ Wrf, const _Float16* __restrict__ Wrb,
    const float* __restrict__ h0f, const float* __restrict__ c0f,
    const float* __restrict__ h0b, const float* __restrict__ c0b,
    float* __restrict__ out) {
  __shared__ __align__(16) _Float16 h_sh[BATCH][UNITS];   // 32 KB
  const int dir = blockIdx.x;
  const _Float16* Z = dir ? Zb : Zf;
  const v16h* Wr = (const v16h*)(dir ? Wrb : Wrf);
  const float* h0 = dir ? h0b : h0f;
  const float* c0 = dir ? c0b : c0f;

  int tid  = threadIdx.x;
  int wave = tid >> 5, lane = tid & 31, half = lane >> 4;
  int mt   = wave & 3;            // batch tile (16 rows)
  int ut0  = (wave >> 2) * 2;     // two u-tiles per wave
  int ucol0 = ut0 * 16 + (lane & 15);

  float creg[2][8];
#pragma unroll
  for (int ui = 0; ui < 2; ++ui) {
    int u = ucol0 + ui * 16;
#pragma unroll
    for (int r = 0; r < 8; ++r) {
      int b = mt * 16 + r + 8 * half;
      creg[ui][r] = c0[b * UNITS + u];
      h_sh[b][u]  = (_Float16)h0[b * UNITS + u];
    }
  }
  __syncthreads();

  const int KC = UNITS / 32;      // 8 k-chunks
  for (int s = 0; s < TLEN; ++s) {
    int t = dir ? (TLEN - 1 - s) : s;
    float hnew[2][8];
#pragma unroll
    for (int ui = 0; ui < 2; ++ui) {
      int ut = ut0 + ui;
      int u  = ucol0 + ui * 16;
      v8f acc0 = {}, acc1 = {}, acc2 = {}, acc3 = {};
      for (int kc = 0; kc < KC; ++kc) {
        v16h a = load_a_frag(&h_sh[mt * 16 + (lane & 15)][0], kc * 32, half);
        const v16h* bp = Wr + (long long)(kc * 64 + ut) * 32 + lane;
        // gates i,f,c,o at n-tiles ut + g*16 -> strides of 16*32 v16h
        v16h b0 = bp[0];
        v16h b1 = bp[512];
        v16h b2 = bp[1024];
        v16h b3 = bp[1536];
        __builtin_prefetch((const void*)(bp + 64 * 32), 0, 3);
        acc0 = wmma16x16x32(a, b0, acc0);
        acc1 = wmma16x16x32(a, b1, acc1);
        acc2 = wmma16x16x32(a, b2, acc2);
        acc3 = wmma16x16x32(a, b3, acc3);
      }
#pragma unroll
      for (int r = 0; r < 8; ++r) {
        int b = mt * 16 + r + 8 * half;
        long long zrow = ((long long)b * TLEN + t) * (4 * UNITS);
        float zi = acc0[r] + (float)Z[zrow + 0 * UNITS + u];
        float zf = acc1[r] + (float)Z[zrow + 1 * UNITS + u];
        float zg = acc2[r] + (float)Z[zrow + 2 * UNITS + u];
        float zo = acc3[r] + (float)Z[zrow + 3 * UNITS + u];
        float ig = 1.f / (1.f + __expf(-zi));
        float fg = 1.f / (1.f + __expf(-zf));
        float gg = tanhf(zg);
        float og = 1.f / (1.f + __expf(-zo));
        float c  = fg * creg[ui][r] + ig * gg;
        creg[ui][r] = c;
        float h = og * tanhf(c);
        hnew[ui][r] = h;
        out[((long long)b * TLEN + t) * (2 * UNITS) + dir * UNITS + u] = h;
      }
    }
    __syncthreads();                         // all h_sh reads done
#pragma unroll
    for (int ui = 0; ui < 2; ++ui) {
      int u = ucol0 + ui * 16;
#pragma unroll
      for (int r = 0; r < 8; ++r) {
        int b = mt * 16 + r + 8 * half;
        h_sh[b][u] = (_Float16)hnew[ui][r];
      }
    }
    __syncthreads();                         // new h visible
  }
}

// ---------------------------------------------------------------------------
// Host launcher
// ---------------------------------------------------------------------------
extern "C" void kernel_launch(void* const* d_in, const int* in_sizes, int n_in,
                              void* d_out, int out_size, void* d_ws, size_t ws_size,
                              hipStream_t stream) {
  const int*   x      = (const int*)d_in[0];
  const float* h0f    = (const float*)d_in[1];
  const float* c0f    = (const float*)d_in[2];
  const float* h0b    = (const float*)d_in[3];
  const float* c0b    = (const float*)d_in[4];
  const float* emb    = (const float*)d_in[5];
  const float* conv_w = (const float*)d_in[6];
  const float* conv_b = (const float*)d_in[7];
  const float* bng    = (const float*)d_in[8];
  const float* bnb    = (const float*)d_in[9];
  const float* bnm    = (const float*)d_in[10];
  const float* bnv    = (const float*)d_in[11];
  const float* wk_f   = (const float*)d_in[12];
  const float* wr_f   = (const float*)d_in[13];
  const float* b_f    = (const float*)d_in[14];
  const float* wk_b   = (const float*)d_in[15];
  const float* wr_b   = (const float*)d_in[16];
  const float* b_b    = (const float*)d_in[17];
  float* out = (float*)d_out;

  char* w = (char*)d_ws;
  size_t off = 0;
  auto alloc = [&](size_t bytes) -> void* {
    void* p = w + off;
    off = (off + bytes + 255) & ~(size_t)255;
    return p;
  };

  _Float16* X0 = (_Float16*)alloc((size_t)ROWS * EMB * 2);
  _Float16* X1 = (_Float16*)alloc((size_t)ROWS * EMB * 2);
  _Float16* Wc[3];
  const size_t convFragHalves = (size_t)80 * 32 * 32 * 16;     // kc*nt*lane*16
  for (int l = 0; l < 3; ++l) Wc[l] = (_Float16*)alloc(convFragHalves * 2);
  float* alpha = (float*)alloc(3 * FILT * 4);
  float* betap = (float*)alloc(3 * FILT * 4);
  const size_t wkFragHalves = (size_t)16 * 64 * 32 * 16;       // K=512, N=1024
  const size_t wrFragHalves = (size_t)8 * 64 * 32 * 16;        // K=256, N=1024
  _Float16* Wkf = (_Float16*)alloc(wkFragHalves * 2);
  _Float16* Wkb = (_Float16*)alloc(wkFragHalves * 2);
  _Float16* Wrf = (_Float16*)alloc(wrFragHalves * 2);
  _Float16* Wrb = (_Float16*)alloc(wrFragHalves * 2);
  _Float16* Zf  = (_Float16*)alloc((size_t)ROWS * 4 * UNITS * 2);
  _Float16* Zb  = (_Float16*)alloc((size_t)ROWS * 4 * UNITS * 2);
  (void)ws_size; (void)in_sizes; (void)n_in; (void)out_size;

  // 1) embed -> f16
  embed_kernel<<<(ROWS * (EMB / 8)) / 256, 256, 0, stream>>>(x, emb, X0);

  // 2) pack weights into B-fragment layout
  for (int l = 0; l < 3; ++l)
    pack_b_kernel<<<(int)((convFragHalves + 255) / 256), 256, 0, stream>>>(
        conv_w + (size_t)l * KW * EMB * FILT, Wc[l], 80, 32, FILT);
  pack_b_kernel<<<(int)((wkFragHalves + 255) / 256), 256, 0, stream>>>(wk_f, Wkf, 16, 64, 4 * UNITS);
  pack_b_kernel<<<(int)((wkFragHalves + 255) / 256), 256, 0, stream>>>(wk_b, Wkb, 16, 64, 4 * UNITS);
  pack_b_kernel<<<(int)((wrFragHalves + 255) / 256), 256, 0, stream>>>(wr_f, Wrf, 8, 64, 4 * UNITS);
  pack_b_kernel<<<(int)((wrFragHalves + 255) / 256), 256, 0, stream>>>(wr_b, Wrb, 8, 64, 4 * UNITS);

  // 3) fold BN + bias
  bnfold_kernel<<<(3 * FILT + 255) / 256, 256, 0, stream>>>(conv_b, bng, bnb, bnm, bnv, alpha, betap);

  // 4) conv stack (ping-pong f16), LDS-staged halo, 32x256 per block
  dim3 convGrid(ROWS / 32, FILT / 256);     // (1024, 2)
  conv_kernel<<<convGrid, 256, 0, stream>>>(X0, Wc[0], alpha + 0 * FILT, betap + 0 * FILT, X1);
  conv_kernel<<<convGrid, 256, 0, stream>>>(X1, Wc[1], alpha + 1 * FILT, betap + 1 * FILT, X0);
  conv_kernel<<<convGrid, 256, 0, stream>>>(X0, Wc[2], alpha + 2 * FILT, betap + 2 * FILT, X1);

  // 5) LSTM input projections (bias folded); 2048 row-tiles x 16 n-groups, one per wave
  const int projBlocks = (2048 * 16) / 8;   // 4096
  proj_kernel<<<projBlocks, 256, 0, stream>>>(X1, Wkf, b_f, Zf);
  proj_kernel<<<projBlocks, 256, 0, stream>>>(X1, Wkb, b_b, Zb);

  // 6) recurrence, fwd + bwd in parallel
  lstm_kernel<<<2, 1024, 0, stream>>>(Zf, Zb, Wrf, Wrb, h0f, c0f, h0b, c0b, out);
}
